// RegionProposalNetwork_72524817760382
// MI455X (gfx1250) — matrix operand order repaired and checked by reading, not compiled, not run on hardware
//
// MI455X (gfx1250) RPN: split-bf16 WMMA implicit-GEMM conv + fused 1x1 heads,
// radix-select top-k + bitonic sort, greedy NMS. Wave32 / WMMA throughout.
// Round 2: B-fragments fully preloaded (single 16-wide load clause per K-iter),
// A-tile software-pipelined through a double-buffered LDS tile (1 barrier/iter).
#include <hip/hip_runtime.h>
#include <math.h>

typedef __bf16 bf16;
typedef bf16  v16bf __attribute__((ext_vector_type(16)));
typedef float v8f   __attribute__((ext_vector_type(8)));

#define NANCH 261888
#define NSEL  6000
#define NPROP 1000
#define KW_ELEMS (9 * 256 * 512)
#define TILES_PER_BATCH 5456   // 87296 pixels / 16

// ---------------------------------------------------------------------------
// Kernel 1: re-pack shared 3x3 conv weights (HWIO fp32) into split-bf16 pair
// with WMMA B-matrix friendly layout: [tap(9)][kchunk(8)][cout(512)][k(32)]
// so each lane's v16bf B-fragment is one contiguous 32-byte load.
// ---------------------------------------------------------------------------
__global__ void rpn_wconv(const float* __restrict__ w,
                          bf16* __restrict__ whi, bf16* __restrict__ wlo) {
  int i = blockIdx.x * blockDim.x + threadIdx.x;
  if (i >= KW_ELEMS) return;
  int co  = i % 512;
  int r   = i / 512;
  int ci  = r % 256;
  int tap = r / 256;
  float x = w[i];
  bf16 h = (bf16)x;
  bf16 l = (bf16)(x - (float)h);
  size_t dst = (((size_t)tap * 8 + (ci >> 5)) * 512 + (size_t)co) * 32 + (ci & 31);
  whi[dst] = h;
  wlo[dst] = l;
}

// ---------------------------------------------------------------------------
// Kernel 2: fused 3x3 conv (implicit GEMM via v_wmma_f32_16x16x32_bf16,
// split-bf16 fp32 emulation) + ReLU + 1x1 cls/box convs + softmax scores.
// One block = 16 pixels x 512 channels; 8 waves each own 64 channels.
// ---------------------------------------------------------------------------
__global__ __launch_bounds__(256)
void rpn_conv_fused(const float* __restrict__ p2, const float* __restrict__ p3,
                    const float* __restrict__ p4, const float* __restrict__ p5,
                    const float* __restrict__ p6,
                    const float* __restrict__ b_shared,
                    const float* __restrict__ w_cls, const float* __restrict__ b_cls,
                    const float* __restrict__ w_box, const float* __restrict__ b_box,
                    const bf16* __restrict__ whi, const bf16* __restrict__ wlo,
                    float* __restrict__ scores, float* __restrict__ deltas) {
  __shared__ bf16  Ahi[2][16][32];       // double-buffered A tile (hi part)
  __shared__ bf16  Alo[2][16][32];       // double-buffered A tile (lo part)
  __shared__ float wst[512 * 18];        // [cout][18]: 6 cls then 12 box
  __shared__ float partW[8][16][18];     // per-wave partials (deterministic)

  const int tid  = threadIdx.x;
  const int lane = tid & 31;
  const int wave = tid >> 5;
  const int half = lane >> 4;
  const int lrow = lane & 15;

  // Decode (batch, level, tile)
  int bid = blockIdx.x;
  int batch = bid / TILES_PER_BATCH;
  int t = bid % TILES_PER_BATCH;
  int lvl, tile;
  if (t < 4096)      { lvl = 0; tile = t; }
  else if (t < 5120) { lvl = 1; tile = t - 4096; }
  else if (t < 5376) { lvl = 2; tile = t - 5120; }
  else if (t < 5440) { lvl = 3; tile = t - 5376; }
  else               { lvl = 4; tile = t - 5440; }
  const int W = 256 >> lvl;
  const float* inp = (lvl == 0) ? p2 : (lvl == 1) ? p3 : (lvl == 2) ? p4
                   : (lvl == 3) ? p5 : p6;
  inp += (size_t)batch * W * W * 256;
  const int m0 = tile * 16;

  // Stage 1x1 head weights into LDS
  for (int i = tid; i < 512 * 18; i += 256) {
    int co = i / 18, j = i % 18;
    wst[i] = (j < 6) ? w_cls[co * 6 + j] : w_box[co * 12 + (j - 6)];
  }

  v8f acc[4];
#pragma unroll
  for (int n = 0; n < 4; ++n)
#pragma unroll
    for (int e = 0; e < 8; ++e) acc[n][e] = 0.f;

  const int p_st  = tid >> 4;           // pixel this thread stages
  const int e2    = tid & 15;           // channel pair
  const int pix_s = m0 + p_st;
  const int h_s = pix_s / W, w_s = pix_s % W;

  // --- A-tile pipeline helpers (kk = tap*8 + kchunk) ---
  auto stage_load = [&](int kk, float& vx, float& vy) {
    const int tap = kk >> 3, kc = kk & 7;
    const int ky = tap / 3 - 1, kx = tap % 3 - 1;
    int hy = h_s + ky, wx = w_s + kx;
    vx = 0.f; vy = 0.f;
    if (hy >= 0 && hy < W && wx >= 0 && wx < W) {
      const float* src = inp + (((size_t)hy * W + wx) << 8) + kc * 32 + e2 * 2;
      vx = src[0]; vy = src[1];
    }
  };
  auto stage_store = [&](int buf, float vx, float vy) {
    bf16 h0 = (bf16)vx, h1 = (bf16)vy;
    Ahi[buf][p_st][e2 * 2]     = h0;
    Ahi[buf][p_st][e2 * 2 + 1] = h1;
    Alo[buf][p_st][e2 * 2]     = (bf16)(vx - (float)h0);
    Alo[buf][p_st][e2 * 2 + 1] = (bf16)(vy - (float)h1);
  };

  {
    float vx, vy;
    stage_load(0, vx, vy);
    stage_store(0, vx, vy);
  }

  for (int kk = 0; kk < 72; ++kk) {
    const int buf = kk & 1;
    __syncthreads();   // tile kk visible in buf; prior reads of buf^1 done

    // A fragments: lane<16 holds K{0..15}; lane>=16 holds K{16..31}
    v16bf ah, al;
#pragma unroll
    for (int e = 0; e < 8; ++e) {
      ah[e]     = Ahi[buf][lrow][half * 16 + e];
      ah[e + 8] = Ahi[buf][lrow][half * 16 + 8 + e];
      al[e]     = Alo[buf][lrow][half * 16 + e];
      al[e + 8] = Alo[buf][lrow][half * 16 + 8 + e];
    }

    // B fragments: preload ALL 8 (hi+lo for 4 sub-tiles) -> one load clause
    const size_t wbase = (size_t)kk * 512 * 32;
    v16bf bh[4], bl[4];
#pragma unroll
    for (int n = 0; n < 4; ++n) {
      const int co = wave * 64 + n * 16 + lrow;
      const size_t off = wbase + (size_t)co * 32 + half * 16;
      bh[n] = *(const v16bf*)(whi + off);
      bl[n] = *(const v16bf*)(wlo + off);
    }

    if (kk + 1 < 72)  // hint next weight block into cache
      __builtin_prefetch(whi + (size_t)(kk + 1) * 512 * 32, 0, 1);

    // Issue next A-tile global loads now; latency hides under the WMMAs
    float nvx = 0.f, nvy = 0.f;
    if (kk + 1 < 72) stage_load(kk + 1, nvx, nvy);

    // split-bf16 fp32 emulation: hi*hi + lo*hi + hi*lo, f32 accumulate.
    // n-innermost order keeps same-accumulator RAWs 4 instructions apart.
#pragma unroll
    for (int n = 0; n < 4; ++n)
      acc[n] = __builtin_amdgcn_wmma_f32_16x16x32_bf16(false, ah, false, bh[n],
                                                       (short)0, acc[n], false, false);
#pragma unroll
    for (int n = 0; n < 4; ++n)
      acc[n] = __builtin_amdgcn_wmma_f32_16x16x32_bf16(false, al, false, bh[n],
                                                       (short)0, acc[n], false, false);
#pragma unroll
    for (int n = 0; n < 4; ++n)
      acc[n] = __builtin_amdgcn_wmma_f32_16x16x32_bf16(false, ah, false, bl[n],
                                                       (short)0, acc[n], false, false);

    // Store next tile into the other buffer (race-free: barrier at loop top)
    if (kk + 1 < 72) stage_store(buf ^ 1, nvx, nvy);
  }
  __syncthreads();

  // Epilogue: bias + ReLU, fused 1x1 heads with deterministic reduction
  float vreg[4][8];
  int cos[4];
#pragma unroll
  for (int n = 0; n < 4; ++n) {
    cos[n] = wave * 64 + n * 16 + lrow;
    float bb = b_shared[cos[n]];
#pragma unroll
    for (int r = 0; r < 8; ++r) vreg[n][r] = fmaxf(acc[n][r] + bb, 0.f);
  }
#pragma unroll
  for (int r = 0; r < 8; ++r) {
    const int M = r + half * 8;                   // pixel index in tile
    for (int j = 0; j < 18; ++j) {
      float s = 0.f;
#pragma unroll
      for (int n = 0; n < 4; ++n) s += vreg[n][r] * wst[cos[n] * 18 + j];
      // reduce across the 16 lanes of this half (same M, disjoint channels)
      s += __shfl_xor(s, 1);
      s += __shfl_xor(s, 2);
      s += __shfl_xor(s, 4);
      s += __shfl_xor(s, 8);
      if (lrow == 0) partW[wave][M][j] = s;
    }
  }
  __syncthreads();

  if (tid < 16) {
    static const int loff[5] = {0, 196608, 245760, 258048, 261120};
    const int pix = m0 + tid;
    const size_t abase = (size_t)batch * NANCH + loff[lvl] + (size_t)pix * 3;
    float lg[6], dl[12];
    for (int j = 0; j < 6; ++j) {
      float s = 0.f;
      for (int w8 = 0; w8 < 8; ++w8) s += partW[w8][tid][j];
      lg[j] = s + b_cls[j];
    }
    for (int j = 0; j < 12; ++j) {
      float s = 0.f;
      for (int w8 = 0; w8 < 8; ++w8) s += partW[w8][tid][6 + j];
      dl[j] = s + b_box[j];
    }
    for (int a = 0; a < 3; ++a) {
      float d = lg[2 * a] - lg[2 * a + 1];         // softmax class-0 prob
      scores[abase + a] = 1.f / (1.f + expf(-d));
      float* dp = deltas + (abase + a) * 4;
      dp[0] = dl[a * 4 + 0]; dp[1] = dl[a * 4 + 1];
      dp[2] = dl[a * 4 + 2]; dp[3] = dl[a * 4 + 3];
    }
  }
}

// ---------------------------------------------------------------------------
// Kernel 3: exact top-6000 per batch. 4-pass MSB radix-select (positive f32
// keys compare as uints), then gather >= threshold and bitonic-sort 8192
// packed (~key,idx) u64 in LDS -> stable descending order (matches jax top_k).
// ---------------------------------------------------------------------------
__global__ __launch_bounds__(1024)
void rpn_topk(const float* __restrict__ scores, float* __restrict__ out_scores,
              int* __restrict__ sel_idx) {
  __shared__ unsigned long long smem[8192];              // 64 KB
  unsigned* hist = (unsigned*)smem;                      // first 1 KB (phase A)
  unsigned* ctl  = ((unsigned*)smem) + 2 * 8190;         // last 2 u64 slots

  const int tid = threadIdx.x;
  const int b   = blockIdx.x;
  const float* sc = scores + (size_t)b * NANCH;

  unsigned pref = 0, need = NSEL;
  for (int pass = 3; pass >= 0; --pass) {
    const int shift = pass * 8;
    const unsigned mask = (pass == 3) ? 0u : (0xFFFFFFFFu << (shift + 8));
    for (int i = tid; i < 256; i += 1024) hist[i] = 0;
    __syncthreads();
    for (int i = tid; i < NANCH; i += 1024) {
      unsigned u = __float_as_uint(sc[i]);
      if ((u & mask) == pref) atomicAdd(&hist[(u >> shift) & 255], 1u);
    }
    __syncthreads();
    if (tid == 0) {
      unsigned cum = 0;
      int v = 255;
      for (; v > 0; --v) {
        if (cum + hist[v] >= need) break;
        cum += hist[v];
      }
      ctl[0] = pref | ((unsigned)v << shift);
      ctl[1] = need - cum;
    }
    __syncthreads();
    pref = ctl[0];
    need = ctl[1];
    __syncthreads();
  }
  const unsigned kstar = pref;                           // 6000th-largest key

  if (tid == 0) ctl[2] = 0;
  __syncthreads();
  for (int i = tid; i < NANCH; i += 1024) {
    unsigned u = __float_as_uint(sc[i]);
    if (u >= kstar) {
      unsigned slot = atomicAdd(&ctl[2], 1u);
      if (slot < 8190u)
        smem[slot] = (((unsigned long long)(~u)) << 32) | (unsigned)i;
    }
  }
  __syncthreads();
  unsigned cnt = ctl[2];
  if (cnt > 8190u) cnt = 8190u;
  __syncthreads();
  for (int i = tid; i < 8192; i += 1024)
    if ((unsigned)i >= cnt) smem[i] = 0xFFFFFFFFFFFFFFFFull;
  __syncthreads();

  // bitonic sort ascending on (~key, idx) -> score desc, idx asc
  for (int k = 2; k <= 8192; k <<= 1) {
    for (int j = k >> 1; j > 0; j >>= 1) {
      for (int i = tid; i < 8192; i += 1024) {
        int ixj = i ^ j;
        if (ixj > i) {
          unsigned long long a = smem[i], c = smem[ixj];
          bool up = ((i & k) == 0);
          if (up ? (a > c) : (a < c)) { smem[i] = c; smem[ixj] = a; }
        }
      }
      __syncthreads();
    }
  }
  for (int q = tid; q < NSEL; q += 1024) {
    unsigned long long v = smem[q];
    unsigned idx = (unsigned)v;
    unsigned key = ~((unsigned)(v >> 32));
    out_scores[b * NSEL + q] = __uint_as_float(key);
    sel_idx[b * NSEL + q]    = (int)idx;
  }
}

// ---------------------------------------------------------------------------
// Kernel 4: anchor regen (exact reference meshgrid order) + delta decode +
// clip; emits top_deltas output and clipped boxes for NMS; zeroes proposals.
// ---------------------------------------------------------------------------
__global__ void rpn_gather(const int* __restrict__ sel_idx,
                           const float* __restrict__ deltas,
                           float* __restrict__ out, float* __restrict__ boxes) {
  int g = blockIdx.x * blockDim.x + threadIdx.x;
  if (g < 2 * NPROP * 4) out[12000 + g] = 0.f;          // zero proposals region
  if (g >= 2 * NSEL) return;
  const int b = g / NSEL;
  const int idx = sel_idx[g];

  static const int loff[5] = {0, 196608, 245760, 258048, 261120};
  int lvl = (idx >= 261120) ? 4 : (idx >= 258048) ? 3 : (idx >= 245760) ? 2
          : (idx >= 196608) ? 1 : 0;
  int q = idx - loff[lvl];
  int H = 256 >> lvl;
  float fs    = (float)(4 << lvl);
  float scale = (float)(32 << lvl);
  int p = q / 3, r = q - p * 3;
  // reference meshgrid('xy') layout: cy varies fastest
  float cy = ((float)(p % H) + 0.5f) * fs;
  float cx = ((float)(p / H) + 0.5f) * fs;
  float sr = (r == 0) ? 0.70710678f : (r == 1) ? 1.0f : 1.41421356f;
  float ah = scale / sr, aw = scale * sr;
  const float inv = 1.0f / 1024.0f;
  float ay1 = (cy - 0.5f * ah) * inv;
  float ax1 = (cx - 0.5f * aw) * inv;
  float ay2 = (cy + 0.5f * ah) * inv;
  float ax2 = (cx + 0.5f * aw) * inv;

  const float* dp = deltas + ((size_t)b * NANCH + idx) * 4;
  float d0 = dp[0], d1 = dp[1], d2 = dp[2], d3 = dp[3];
  float* td = out + 20000 + (size_t)g * 4;               // top_deltas output
  td[0] = d0; td[1] = d1; td[2] = d2; td[3] = d3;

  float hh = ay2 - ay1, wd = ax2 - ax1;
  float ncy = ay1 + 0.5f * hh + d0 * hh;
  float ncx = ax1 + 0.5f * wd + d1 * wd;
  hh *= expf(d2);
  wd *= expf(d3);
  float ry1 = ncy - 0.5f * hh, rx1 = ncx - 0.5f * wd;
  float ry2 = ry1 + hh,        rx2 = rx1 + wd;
  float* bo = boxes + (size_t)g * 4;
  bo[0] = fminf(fmaxf(ry1, 0.f), 1.f);
  bo[1] = fminf(fmaxf(rx1, 0.f), 1.f);
  bo[2] = fminf(fmaxf(ry2, 0.f), 1.f);
  bo[3] = fminf(fmaxf(rx2, 0.f), 1.f);
}

// ---------------------------------------------------------------------------
// Kernel 5: greedy NMS (exact reference scan semantics), boxes in registers,
// one workgroup per batch, early exit at 1000 kept proposals.
// ---------------------------------------------------------------------------
__global__ __launch_bounds__(1024)
void rpn_nms(const float* __restrict__ boxes, float* __restrict__ proposals) {
  const int b = blockIdx.x, tid = threadIdx.x;
  const float* bx = boxes + (size_t)b * NSEL * 4;
  float y1[6], x1[6], y2[6], x2[6], ar[6];
  bool act[6];
#pragma unroll
  for (int m = 0; m < 6; ++m) {
    int j = tid + m * 1024;
    act[m] = false;
    y1[m] = x1[m] = y2[m] = x2[m] = ar[m] = 0.f;
    if (j < NSEL) {
      y1[m] = bx[j * 4 + 0]; x1[m] = bx[j * 4 + 1];
      y2[m] = bx[j * 4 + 2]; x2[m] = bx[j * 4 + 3];
      ar[m] = (y2[m] - y1[m]) * (x2[m] - x1[m]);
      act[m] = true;
    }
  }
  __shared__ float cb[5];
  __shared__ int csel;
  __shared__ int ccount;
  if (tid == 0) ccount = 0;
  __syncthreads();
  for (int i = 0; i < NSEL; ++i) {
    if (tid == (i & 1023)) {
      int m = i >> 10;
      csel = act[m] ? 1 : 0;
      cb[0] = y1[m]; cb[1] = x1[m]; cb[2] = y2[m]; cb[3] = x2[m]; cb[4] = ar[m];
    }
    __syncthreads();
    if (csel) {
      if (tid == 0) {
        int rk = ccount;
        if (rk < NPROP) {
          float* pp = proposals + ((size_t)b * NPROP + rk) * 4;
          pp[0] = cb[0]; pp[1] = cb[1]; pp[2] = cb[2]; pp[3] = cb[3];
        }
        ccount = rk + 1;
      }
#pragma unroll
      for (int m = 0; m < 6; ++m) {
        int j = tid + m * 1024;
        if (j > i && j < NSEL && act[m]) {
          float iy1 = fmaxf(y1[m], cb[0]);
          float ix1 = fmaxf(x1[m], cb[1]);
          float iy2 = fminf(y2[m], cb[2]);
          float ix2 = fminf(x2[m], cb[3]);
          float inter = fmaxf(iy2 - iy1, 0.f) * fmaxf(ix2 - ix1, 0.f);
          float iou = inter / (ar[m] + cb[4] - inter + 1e-8f);
          if (iou > 0.8f) act[m] = false;
        }
      }
    }
    __syncthreads();
    if (ccount >= NPROP) break;
  }
}

// ---------------------------------------------------------------------------
extern "C" void kernel_launch(void* const* d_in, const int* in_sizes, int n_in,
                              void* d_out, int out_size, void* d_ws, size_t ws_size,
                              hipStream_t stream) {
  (void)in_sizes; (void)n_in; (void)out_size; (void)ws_size;
  const float* p2       = (const float*)d_in[0];
  const float* p3       = (const float*)d_in[1];
  const float* p4       = (const float*)d_in[2];
  const float* p5       = (const float*)d_in[3];
  const float* p6       = (const float*)d_in[4];
  const float* w_shared = (const float*)d_in[5];
  const float* b_shared = (const float*)d_in[6];
  const float* w_cls    = (const float*)d_in[7];
  const float* b_cls    = (const float*)d_in[8];
  const float* w_box    = (const float*)d_in[9];
  const float* b_box    = (const float*)d_in[10];
  float* out = (float*)d_out;   // [0,12000) scores | [12000,20000) proposals | [20000,68000) deltas

  // workspace layout (all offsets 128B aligned; total ~23.8 MB)
  char* ws = (char*)d_ws;
  bf16*  whi     = (bf16*)(ws + 0);          // 2,359,296 B
  bf16*  wlo     = (bf16*)(ws + 2359296);    // 2,359,296 B
  float* scoresB = (float*)(ws + 4718592);   // 2,095,104 B
  float* deltasB = (float*)(ws + 6813696);   // 16,760,832 B
  int*   sel_idx = (int*)  (ws + 23574528);  // 48,000 B
  float* boxes   = (float*)(ws + 23622528);  // 192,000 B

  rpn_wconv<<<dim3((KW_ELEMS + 255) / 256), dim3(256), 0, stream>>>(w_shared, whi, wlo);
  rpn_conv_fused<<<dim3(2 * TILES_PER_BATCH), dim3(256), 0, stream>>>(
      p2, p3, p4, p5, p6, b_shared, w_cls, b_cls, w_box, b_box,
      whi, wlo, scoresB, deltasB);
  rpn_topk<<<dim3(2), dim3(1024), 0, stream>>>(scoresB, out, sel_idx);
  rpn_gather<<<dim3(47), dim3(256), 0, stream>>>(sel_idx, deltasB, out, boxes);
  rpn_nms<<<dim3(2), dim3(1024), 0, stream>>>(boxes, out + 12000);
}
